// Deform_12987981103531
// MI455X (gfx1250) — matrix operand order repaired and checked by reading, not compiled, not run on hardware
//
#include <hip/hip_runtime.h>

typedef __attribute__((ext_vector_type(16))) _Float16     v16h;
typedef __attribute__((ext_vector_type(8)))  float        v8f;
typedef __attribute__((ext_vector_type(4)))  unsigned int v4u;   // POD 16B vector

// explicit global-address-space pointers so hot loads/stores lower to
// global_load_*/global_store_* (flat_* also ties up DScnt / LDS arbitration)
typedef __attribute__((address_space(1))) const v4u*      gcq4;
typedef __attribute__((address_space(1)))       v4u*      gq4;
typedef __attribute__((address_space(1))) const float*    gcf;
typedef __attribute__((address_space(1)))       float*    gf;
typedef __attribute__((address_space(1)))       _Float16* gh;

#define B_    4
#define C_    256
#define H_    64
#define W_    64
#define O_    256
#define HW_   4096
#define K2_   9
#define KDIM_ 2304          // C_ * 9
#define NSLAB 72            // KDIM_ / 32 k-slabs
#define NMT   256           // HW_ / 16 m-tiles per batch

// workspace layout (bytes):
//   off  : float    [B][18][HW]                     @ 0          (1,179,648)
//   wBF  : f16 frag [72 kslab][16 ntile][32 ln][16] @ 1,179,648  (1,179,648)
//   colsF: f16 frag [B][256 mtile][72 kslab][32][16]@ 2,359,296  (75,497,472)
#define WS_OFF_WB   1179648u
#define WS_OFF_COLS 2359296u

__device__ __forceinline__ int iclamp(int v, int lo, int hi) {
    return v < lo ? lo : (v > hi ? hi : v);
}

// ---------------------------------------------------------------------------
// Kernel 1: offset conv (3x3, pad=1, dil=1), C=256 -> 18, clipped to [-1,1]
// ---------------------------------------------------------------------------
__global__ void __launch_bounds__(256) offset_conv_kernel(
    const float* __restrict__ x, const float* __restrict__ ow,
    const float* __restrict__ ob, float* __restrict__ off) {
    const int tid = threadIdx.x;
    const int w   = tid & 63;
    const int hq  = tid >> 6;
    const int hb  = blockIdx.x & 15;
    const int bko = blockIdx.x >> 4;
    const int ko  = bko % 18;
    const int b   = bko / 18;
    const int h   = hb * 4 + hq;

    float acc = ob[ko];
    gcf xb = (gcf)(x  + (size_t)b  * C_ * HW_);
    gcf wk = (gcf)(ow + (size_t)ko * C_ * 9);
    for (int c = 0; c < C_; ++c) {
        gcf xc = xb + c * HW_;
        gcf wc = wk + c * 9;
        #pragma unroll
        for (int dh = 0; dh < 3; ++dh) {
            const int y = h + dh - 1;
            if (y < 0 || y >= H_) continue;
            #pragma unroll
            for (int dw = 0; dw < 3; ++dw) {
                const int xw = w + dw - 1;
                if (xw < 0 || xw >= W_) continue;
                acc = fmaf(xc[y * 64 + xw], wc[dh * 3 + dw], acc);
            }
        }
    }
    acc = fminf(1.0f, fmaxf(-1.0f, acc));
    ((gf)off)[(size_t)(b * 18 + ko) * HW_ + h * 64 + w] = acc;
}

// ---------------------------------------------------------------------------
// Kernel 2: deform_w f32 (O,C,9) -> B-fragment-order f16 (k' = n*256 + c),
// plus passthrough copy of deform_w (second reference output).
// B frag map: element (k,n_local) -> lane = n_local + 16*(kin>>4), slot = kin&15
// ---------------------------------------------------------------------------
__global__ void __launch_bounds__(256) wcvt_kernel(
    const float* __restrict__ dw, _Float16* __restrict__ wBF,
    float* __restrict__ wcopy) {
    const int idx = blockIdx.x * 256 + threadIdx.x;   // 0 .. 589823
    const int kp  = idx >> 8;        // k' = n*256 + c
    const int o   = idx & 255;
    const int n   = kp >> 8;
    const int c   = kp & 255;
    const float v = ((gcf)dw)[(size_t)o * KDIM_ + c * 9 + n];

    const int kslab = kp >> 5;
    const int kin   = kp & 31;
    const int lane  = (o & 15) + 16 * (kin >> 4);
    const int slot  = kin & 15;
    ((gh)wBF)[((size_t)(kslab * 16 + (o >> 4)) * 32 + lane) * 16 + slot] = (_Float16)v;
    ((gf)wcopy)[idx] = ((gcf)dw)[idx];
}

// ---------------------------------------------------------------------------
// Kernel 3: bilinear deformable sampling, written directly in WMMA A-fragment
// order. For fixed (pixel, n), a 32-c chunk fills exactly the lane-m and
// lane-(m+16) 32B chunks of one 16x32 fragment -> four 16B stores.
// block: 256 threads = 16 m-tiles x 16 pixels; grid: B*(HW/256)
// ---------------------------------------------------------------------------
__global__ void __launch_bounds__(256) sample_kernel(
    const float* __restrict__ x, const float* __restrict__ off,
    _Float16* __restrict__ colsF) {
    const int tid = threadIdx.x;
    const int b   = blockIdx.x >> 4;
    const int hw  = ((blockIdx.x & 15) << 8) + tid;
    const int h   = hw >> 6;
    const int w   = hw & 63;
    const int mtile = hw >> 4;     // global m-tile within batch (0..255)
    const int ml    = hw & 15;     // m_local within tile
    gcf offb = (gcf)(off + (size_t)b * 18 * HW_ + hw);

    int   i00[9], i01[9], i10[9], i11[9];
    float f00[9], f01[9], f10[9], f11[9];
    #pragma unroll
    for (int n = 0; n < 9; ++n) {
        const float dy = offb[(2 * n + 0) * HW_];
        const float dx = offb[(2 * n + 1) * HW_];
        const float py = dy + (float)(-2 + (n / 3) * 2 + h);
        const float px = dx + (float)(-2 + (n % 3) * 2 + w);
        const float y0f = floorf(py), x0f = floorf(px);
        const float wy1 = py - y0f, wx1 = px - x0f;
        const float wy0 = 1.0f - wy1, wx0 = 1.0f - wx1;
        const int y0 = (int)y0f, x0 = (int)x0f;
        const int y1 = y0 + 1,   x1 = x0 + 1;
        const bool vy0 = (y0 >= 0) & (y0 < H_);
        const bool vy1 = (y1 >= 0) & (y1 < H_);
        const bool vx0 = (x0 >= 0) & (x0 < W_);
        const bool vx1 = (x1 >= 0) & (x1 < W_);
        const int cy0 = iclamp(y0, 0, H_ - 1), cy1 = iclamp(y1, 0, H_ - 1);
        const int cx0 = iclamp(x0, 0, W_ - 1), cx1 = iclamp(x1, 0, W_ - 1);
        i00[n] = cy0 * 64 + cx0;  i01[n] = cy0 * 64 + cx1;
        i10[n] = cy1 * 64 + cx0;  i11[n] = cy1 * 64 + cx1;
        f00[n] = (vy0 && vx0) ? wy0 * wx0 : 0.0f;
        f01[n] = (vy0 && vx1) ? wy0 * wx1 : 0.0f;
        f10[n] = (vy1 && vx0) ? wy1 * wx0 : 0.0f;
        f11[n] = (vy1 && vx1) ? wy1 * wx1 : 0.0f;
    }

    gcf xb = (gcf)(x + (size_t)b * C_ * HW_);
    gq4 cf = (gq4)colsF;
    const size_t tilebase0 = ((size_t)b * NMT + mtile) * NSLAB;  // in slabs

    union Pack { _Float16 h[8]; v4u q; };

    for (int n = 0; n < 9; ++n) {
        const int  a00 = i00[n], a01 = i01[n], a10 = i10[n], a11 = i11[n];
        const float g00 = f00[n], g01 = f01[n], g10 = f10[n], g11 = f11[n];
        for (int cc = 0; cc < C_; cc += 32) {
            Pack p00, p01, p10, p11;   // lane-m: k 0..15 ; lane-m+16: 16..31
            #pragma unroll
            for (int j = 0; j < 32; ++j) {
                gcf xc = xb + (size_t)(cc + j) * HW_;
                const float v = g00 * xc[a00] + g01 * xc[a01]
                              + g10 * xc[a10] + g11 * xc[a11];
                const _Float16 hv = (_Float16)v;
                if      (j <  8) p00.h[j]      = hv;
                else if (j < 16) p10.h[j - 8]  = hv;
                else if (j < 24) p01.h[j - 16] = hv;
                else             p11.h[j - 24] = hv;
            }
            const int kslab = n * 8 + (cc >> 5);
            const size_t u4base = (tilebase0 + kslab) * 64;  // 64 x 16B per frag
            cf[u4base + ml * 2 + 0]        = p00.q;
            cf[u4base + ml * 2 + 1]        = p10.q;
            cf[u4base + (ml + 16) * 2 + 0] = p01.q;
            cf[u4base + (ml + 16) * 2 + 1] = p11.q;
        }
    }
}

// ---------------------------------------------------------------------------
// Kernel 4: WMMA GEMM, fragments loaded straight from global with explicit
// double buffering: slab k+1 loads issue before slab-k WMMAs, so the per-
// iteration wait only covers loads already a full iteration old.
// Block tile 128(M) x 64(N); 8 waves, each wave 32x32 (2x2 WMMA 16x16x32).
// ---------------------------------------------------------------------------
#define MT 128
#define NT 64
#define CO_STRIDE 132     // 128 + 4 f32 pad

__global__ void __launch_bounds__(256) gemm_kernel(
    const _Float16* __restrict__ colsF, const _Float16* __restrict__ wBF,
    const float* __restrict__ bias, float* __restrict__ out) {
    __shared__ __align__(16) float Co[64 * CO_STRIDE];  // epilogue transpose

    const int tid = threadIdx.x;
    int bid = blockIdx.x;
    const int nt = bid & 3;  bid >>= 2;
    const int mt = bid & 31; bid >>= 5;
    const int b  = bid;
    const int m0 = mt * MT;
    const int n0 = nt * NT;

    const int lane = tid & 31;
    const int wave = tid >> 5;
    const int wm   = wave & 3;    // m sub-tile (32 rows = 2 m-tiles)
    const int wn   = wave >> 2;   // n sub-tile (32 cols = 2 n-tiles)
    const int lrow = lane & 15;
    const int half = lane >> 4;

    v8f zero = {};
    v8f acc[2][2];
    #pragma unroll
    for (int i = 0; i < 2; ++i)
        #pragma unroll
        for (int j = 0; j < 2; ++j) acc[i][j] = zero;

    // fragment stream pointers (16B units; one frag = 64 x 16B)
    gcq4 cfv = (gcq4)(const v4u*)colsF;
    gcq4 wbv = (gcq4)(const v4u*)wBF;
    gcq4 a0 = cfv + ((size_t)b * NMT + (mt * 8 + wm * 2 + 0)) * NSLAB * 64 + lane * 2;
    gcq4 a1 = cfv + ((size_t)b * NMT + (mt * 8 + wm * 2 + 1)) * NSLAB * 64 + lane * 2;
    gcq4 b0 = wbv + (size_t)(nt * 4 + wn * 2 + 0) * 64 + lane * 2;
    gcq4 b1 = wbv + (size_t)(nt * 4 + wn * 2 + 1) * 64 + lane * 2;

    union AF { v16h v; v4u q[2]; };
    AF af[2][2], bf[2][2];        // [buffer][fragment]

    // prime buffer 0 (slab 0)
    af[0][0].q[0] = a0[0]; af[0][0].q[1] = a0[1]; a0 += 64;
    af[0][1].q[0] = a1[0]; af[0][1].q[1] = a1[1]; a1 += 64;
    bf[0][0].q[0] = b0[0]; bf[0][0].q[1] = b0[1]; b0 += 16 * 64;
    bf[0][1].q[0] = b1[0]; bf[0][1].q[1] = b1[1]; b1 += 16 * 64;

    #pragma unroll 2
    for (int ks = 0; ks < NSLAB - 1; ++ks) {
        const int cur = ks & 1;
        const int nxt = cur ^ 1;
        // issue next slab's loads first
        af[nxt][0].q[0] = a0[0]; af[nxt][0].q[1] = a0[1]; a0 += 64;
        af[nxt][1].q[0] = a1[0]; af[nxt][1].q[1] = a1[1]; a1 += 64;
        bf[nxt][0].q[0] = b0[0]; bf[nxt][0].q[1] = b0[1]; b0 += 16 * 64;
        bf[nxt][1].q[0] = b1[0]; bf[nxt][1].q[1] = b1[1]; b1 += 16 * 64;
        // compute current slab
        #pragma unroll
        for (int sm = 0; sm < 2; ++sm)
            #pragma unroll
            for (int sn = 0; sn < 2; ++sn)
                acc[sm][sn] = __builtin_amdgcn_wmma_f32_16x16x32_f16(
                    false, af[cur][sm].v, false, bf[cur][sn].v,
                    (short)0, acc[sm][sn], false, false);
    }
    {   // final slab (NSLAB-1 = 71 -> buffer 1)
        const int cur = (NSLAB - 1) & 1;
        #pragma unroll
        for (int sm = 0; sm < 2; ++sm)
            #pragma unroll
            for (int sn = 0; sn < 2; ++sn)
                acc[sm][sn] = __builtin_amdgcn_wmma_f32_16x16x32_f16(
                    false, af[cur][sm].v, false, bf[cur][sn].v,
                    (short)0, acc[sm][sn], false, false);
    }

    // ---- epilogue: transpose through LDS so stores are contiguous in hw ----
    #pragma unroll
    for (int sm = 0; sm < 2; ++sm)
        #pragma unroll
        for (int sn = 0; sn < 2; ++sn) {
            const int ocol  = wn * 32 + sn * 16 + lrow;
            const int mbase = wm * 32 + sm * 16 + half * 8;
            #pragma unroll
            for (int r = 0; r < 8; ++r)
                Co[ocol * CO_STRIDE + mbase + r] = acc[sm][sn][r];
        }
    __syncthreads();
    {
        const int ol  = tid >> 2;   // 0..63 local output channel
        const int seg = tid & 3;    // 32-float segment
        const float bv = ((gcf)bias)[n0 + ol];
        gf op = (gf)(out + (size_t)(b * O_ + n0 + ol) * HW_ + m0 + seg * 32);
        #pragma unroll
        for (int j = 0; j < 32; ++j)
            op[j] = Co[ol * CO_STRIDE + seg * 32 + j] + bv;
    }
}

// ---------------------------------------------------------------------------
extern "C" void kernel_launch(void* const* d_in, const int* in_sizes, int n_in,
                              void* d_out, int out_size, void* d_ws, size_t ws_size,
                              hipStream_t stream) {
    const float* x  = (const float*)d_in[0];
    const float* ow = (const float*)d_in[1];
    const float* ob = (const float*)d_in[2];
    const float* dw = (const float*)d_in[3];
    const float* db = (const float*)d_in[4];

    float* out   = (float*)d_out;
    float* wcopy = out + (size_t)B_ * O_ * HW_;   // second output: deform_w

    unsigned char* ws = (unsigned char*)d_ws;
    float*    off   = (float*)ws;
    _Float16* wBF   = (_Float16*)(ws + WS_OFF_WB);
    _Float16* colsF = (_Float16*)(ws + WS_OFF_COLS);

    offset_conv_kernel<<<B_ * 18 * (H_ / 4), 256, 0, stream>>>(x, ow, ob, off);
    wcvt_kernel<<<(KDIM_ * O_) / 256, 256, 0, stream>>>(dw, wBF, wcopy);
    sample_kernel<<<B_ * (HW_ / 256), 256, 0, stream>>>(x, off, colsF);
    gemm_kernel<<<B_ * (HW_ / MT) * (O_ / NT), 256, 0, stream>>>(colsF, wBF, db, out);
}
